// Block_38663295599425
// MI455X (gfx1250) — compile-verified, matrix operand order checked by reading, not run on hardware
//
#include <hip/hip_runtime.h>
#include <hip/hip_bf16.h>

// ---------------- problem constants ----------------
#define BB 2
#define TT 2048
#define DD 1024
#define HH 16
#define HS 64
#define BT (BB * TT)       // 4096 rows
#define DFF (4 * DD)       // 4096
#define PAD 56             // LDS row stride in halves: 112B -> 16B aligned, 16 distinct banks

typedef __attribute__((ext_vector_type(16))) _Float16 v16h;
typedef __attribute__((ext_vector_type(8)))  float    v8f;
typedef __attribute__((ext_vector_type(4)))  int      v4i;

// ---------------- optional CDNA5 async global->LDS path ----------------
#if defined(__has_builtin)
# if __has_builtin(__builtin_amdgcn_global_load_async_to_lds_b128)
#  define HAVE_ASYNC_LDS 1
# endif
# if __has_builtin(__builtin_amdgcn_s_wait_asynccnt)
#  define HAVE_WAIT_ASYNC 1
# endif
#endif

#if defined(HAVE_ASYNC_LDS)
typedef __attribute__((address_space(1))) v4i gv4i_t;  // global int4
typedef __attribute__((address_space(3))) v4i lv4i_t;  // LDS int4
__device__ __forceinline__ void async_cp16(void* lds, const void* g) {
  // copies 16 bytes per lane, tracked by ASYNCcnt
  __builtin_amdgcn_global_load_async_to_lds_b128(
      (gv4i_t*)(void*)g, (lv4i_t*)lds, 0, 0);
}
template <int N>
__device__ __forceinline__ void wait_async() {
# if defined(HAVE_WAIT_ASYNC)
  __builtin_amdgcn_s_wait_asynccnt((unsigned short)N);
# else
  if (N == 0) asm volatile("s_wait_asynccnt 0" ::: "memory");
  else        asm volatile("s_wait_asynccnt 4" ::: "memory");
# endif
}
#endif

// ---------------- WMMA helpers ----------------
__device__ __forceinline__ v8f wmma16(v16h a, v16h b, v8f c) {
  // D = A(16x32 f16) * B(32x16 f16) + C(16x16 f32)
  return __builtin_amdgcn_wmma_f32_16x16x32_f16(false, a, false, b, (short)0, c,
                                                false, false);
}

// A-fragment loader (16-bit A 16x32 layout):
//   lane<16 : row M = lane,    holds K = 0..7  and 16..23
//   lane>=16: row M = lane-16, holds K = 8..15 and 24..31
// B-fragments (stored row=N, col=K) use the same pattern with N as the row.
__device__ __forceinline__ v16h load_frag(const _Float16* p, int stride) {
  int l = threadIdx.x & 31;
  const _Float16* r = p + (size_t)(l & 15) * stride + ((l >> 4) << 3);
  v16h f;
  ((uint4*)&f)[0] = *(const uint4*)(r);
  ((uint4*)&f)[1] = *(const uint4*)(r + 16);
  return f;
}

__device__ __forceinline__ _Float16 to_h(float x) { return (_Float16)x; }

// ---------------- weight packing: f32 -> f16, ALL weights TRANSPOSED --------
// Every packed weight is stored [N][K] so GEMM B-tiles stage with contiguous
// 16B rows (identical to A-tiles, no in-loop transpose).
__global__ __launch_bounds__(256) void pack_weights(
    const float* __restrict__ Wq, const float* __restrict__ Wk,
    const float* __restrict__ Wv, const float* __restrict__ Wp,
    const float* __restrict__ W1, const float* __restrict__ W2,
    _Float16* __restrict__ wqkvT, _Float16* __restrict__ wpT,
    _Float16* __restrict__ w1T, _Float16* __restrict__ w2T) {
  long i = (long)blockIdx.x * 256 + threadIdx.x;
  const long N_QKV = (long)3 * DD * DD;       // [3D][D]
  const long N_P   = (long)DD * DD;           // [D][D]
  const long N_W1  = (long)DFF * DD;          // [4D][D]
  const long N_W2  = (long)DD * DFF;          // [D][4D]
  if (i < N_QKV) {
    int n = (int)(i / DD), d = (int)(i % DD);
    int sel = n >> 10, w = n & 1023, h = w >> 6, e = w & 63;
    const float* W = (sel == 0) ? Wq : (sel == 1) ? Wk : Wv;
    wqkvT[i] = to_h(W[((size_t)h * DD + d) * HS + e]);
    return;
  }
  i -= N_QKV;
  if (i < N_P) {
    int n = (int)(i / DD), d = (int)(i % DD);
    wpT[i] = to_h(Wp[(size_t)d * DD + n]);
    return;
  }
  i -= N_P;
  if (i < N_W1) {
    int n = (int)(i / DD), d = (int)(i % DD);
    w1T[i] = to_h(W1[(size_t)d * DFF + n]);
    return;
  }
  i -= N_W1;
  if (i < N_W2) {
    int n = (int)(i / DFF), d = (int)(i % DFF);
    w2T[i] = to_h(W2[(size_t)d * DD + n]);
    return;
  }
}

// ---------------- LayerNorm: writes f32 (residual) + f16 (GEMM input) -------
__global__ __launch_bounds__(256) void ln_kernel(
    const float* __restrict__ x, const float* __restrict__ g,
    const float* __restrict__ b, float* __restrict__ yf,
    _Float16* __restrict__ yh) {
  __shared__ float s1[256], s2[256];
  int row = blockIdx.x, tid = threadIdx.x;
  const float* xr = x + (size_t)row * DD;
  float4 xv = ((const float4*)xr)[tid];          // 4 elements / thread
  float sum = xv.x + xv.y + xv.z + xv.w;
  float sq  = xv.x * xv.x + xv.y * xv.y + xv.z * xv.z + xv.w * xv.w;
  s1[tid] = sum; s2[tid] = sq;
  __syncthreads();
  for (int off = 128; off; off >>= 1) {
    if (tid < off) { s1[tid] += s1[tid + off]; s2[tid] += s2[tid + off]; }
    __syncthreads();
  }
  float mu  = s1[0] * (1.0f / DD);
  float var = s2[0] * (1.0f / DD) - mu * mu;
  float rs  = rsqrtf(var + 1e-5f);
  float vals[4] = {xv.x, xv.y, xv.z, xv.w};
  for (int j = 0; j < 4; ++j) {
    int c = tid * 4 + j;
    float xn = (vals[j] - mu) * rs * g[c] + b[c];
    yf[(size_t)row * DD + c] = xn;
    yh[(size_t)row * DD + c] = to_h(xn);
  }
}

// ---------------- generic 128x128 WMMA GEMM with fused epilogues ------------
#define MODE_QKV  0
#define MODE_PROJ 1
#define MODE_FFN1 2
#define MODE_FFN2 3

__global__ __launch_bounds__(256) void gemm128(
    const _Float16* __restrict__ A,   // [M][K] row-major f16
    const _Float16* __restrict__ Bt,  // [N][K] row-major f16 (pre-transposed)
    int N, int K, int mode,
    const float* __restrict__ bias, const float* __restrict__ resid,
    float* __restrict__ outf, _Float16* __restrict__ outh,
    _Float16* __restrict__ qp, _Float16* __restrict__ kp,
    _Float16* __restrict__ vp) {
#if defined(HAVE_ASYNC_LDS)
  __shared__ _Float16 As[2][128 * PAD];
  __shared__ _Float16 Bs[2][128 * PAD];
#else
  __shared__ _Float16 As[1][128 * PAD];
  __shared__ _Float16 Bs[1][128 * PAD];
#endif
  const int tid = threadIdx.x;
  const int m0 = blockIdx.y * 128;
  const int n0 = blockIdx.x * 128;
  const int wave = tid >> 5, l = tid & 31;
  const int wm = (wave & 1) * 64;   // wave M offset in tile
  const int wn = (wave >> 1) * 32;  // wave N offset in tile
  const int r = tid >> 2, c = (tid & 3) * 8;  // staging coords (64 rows/pass)

  v8f acc[4][2];
  for (int i = 0; i < 4; ++i)
    for (int j = 0; j < 2; ++j) acc[i][j] = (v8f){};

#if defined(HAVE_ASYNC_LDS)
  auto stage = [&](int kk, int p) {
    async_cp16(&As[p][(size_t)r * PAD + c],        A  + (size_t)(m0 + r) * K + kk + c);
    async_cp16(&As[p][(size_t)(r + 64) * PAD + c], A  + (size_t)(m0 + r + 64) * K + kk + c);
    async_cp16(&Bs[p][(size_t)r * PAD + c],        Bt + (size_t)(n0 + r) * K + kk + c);
    async_cp16(&Bs[p][(size_t)(r + 64) * PAD + c], Bt + (size_t)(n0 + r + 64) * K + kk + c);
  };
  int p = 0;
  stage(0, 0);
#endif

  for (int kk = 0; kk < K; kk += 32) {
#if defined(HAVE_ASYNC_LDS)
    if (kk + 32 < K) { stage(kk + 32, p ^ 1); wait_async<4>(); }
    else             { wait_async<0>(); }
    __syncthreads();
    const _Float16* as = As[p];
    const _Float16* bs = Bs[p];
#else
    *(uint4*)&As[0][(size_t)r * PAD + c] =
        *(const uint4*)(A + (size_t)(m0 + r) * K + kk + c);
    *(uint4*)&As[0][(size_t)(r + 64) * PAD + c] =
        *(const uint4*)(A + (size_t)(m0 + r + 64) * K + kk + c);
    *(uint4*)&Bs[0][(size_t)r * PAD + c] =
        *(const uint4*)(Bt + (size_t)(n0 + r) * K + kk + c);
    *(uint4*)&Bs[0][(size_t)(r + 64) * PAD + c] =
        *(const uint4*)(Bt + (size_t)(n0 + r + 64) * K + kk + c);
    __syncthreads();
    const _Float16* as = As[0];
    const _Float16* bs = Bs[0];
#endif
    v16h af[4], bf[2];
    for (int i = 0; i < 4; ++i) af[i] = load_frag(as + (wm + 16 * i) * PAD, PAD);
    for (int j = 0; j < 2; ++j) bf[j] = load_frag(bs + (wn + 16 * j) * PAD, PAD);
    for (int i = 0; i < 4; ++i)
      for (int j = 0; j < 2; ++j) acc[i][j] = wmma16(af[i], bf[j], acc[i][j]);
    __syncthreads();
#if defined(HAVE_ASYNC_LDS)
    p ^= 1;
#endif
  }

  // epilogue: C lane map -> col = base + (l&15); rows = base + r + (l>=16)*8
  const int cn = l & 15, rb = (l >> 4) * 8;
  for (int i = 0; i < 4; ++i)
    for (int j = 0; j < 2; ++j)
      for (int rr = 0; rr < 8; ++rr) {
        int gm = m0 + wm + 16 * i + rb + rr;
        int gn = n0 + wn + 16 * j + cn;
        float v = acc[i][j][rr];
        if (mode == MODE_QKV) {
          int sel = gn >> 10, w = gn & 1023, h = w >> 6, e = w & 63;
          int bi = gm >> 11, t = gm & (TT - 1);
          size_t dst = (((size_t)(bi * HH + h)) * TT + t) * HS + e;
          if (sel == 0)       qp[dst] = to_h(v * 0.03125f);  // fold D^-0.5
          else if (sel == 1)  kp[dst] = to_h(v);
          else                vp[dst] = to_h(v);
        } else if (mode == MODE_PROJ) {
          v += bias[gn] + resid[(size_t)gm * DD + gn];
          outf[(size_t)gm * DD + gn] = v;
        } else if (mode == MODE_FFN1) {
          v += bias[gn];
          v = fmaxf(v, 0.0f);
          outh[(size_t)gm * DFF + gn] = to_h(v);
        } else {  // MODE_FFN2
          v += bias[gn] + resid[(size_t)gm * DD + gn];
          outf[(size_t)gm * DD + gn] = v;
        }
      }
}

// ---------------- attention pass 1: per-key-column max / sumexp -------------
// softmax is over the QUERY axis (per reference): m_k, Z_k over q >= k.
__global__ __launch_bounds__(256) void attn_pass1(
    const _Float16* __restrict__ q, const _Float16* __restrict__ k,
    float* __restrict__ cmax, float* __restrict__ csum) {
  const int bh = blockIdx.y;
  const int wave = threadIdx.x >> 5, l = threadIdx.x & 31;
  const int k0 = blockIdx.x * 128 + wave * 16;   // this wave's 16 key columns
  const _Float16* qh = q + (size_t)bh * TT * HS;
  const _Float16* kh = k + (size_t)bh * TT * HS;

  // K^T fragments for this column strip (lane row = key token)
  v16h kf0 = load_frag(kh + (size_t)k0 * HS, HS);
  v16h kf1 = load_frag(kh + (size_t)k0 * HS + 32, HS);

  float m = -INFINITY, s = 0.0f;
  const int kcol = k0 + (l & 15);
  for (int qt = k0; qt < TT; qt += 16) {
    v16h qa0 = load_frag(qh + (size_t)qt * HS, HS);
    v16h qa1 = load_frag(qh + (size_t)qt * HS + 32, HS);
    v8f c = (v8f){};
    c = wmma16(qa0, kf0, c);
    c = wmma16(qa1, kf1, c);
    const int qrb = qt + ((l >> 4) ? 8 : 0);
    for (int r = 0; r < 8; ++r) {
      int qi = qrb + r;
      if (qi >= kcol) {               // causal (q >= k valid)
        float val = c[r];
        if (val > m) { s = s * __expf(m - val) + 1.0f; m = val; }
        else         { s += __expf(val - m); }
      }
    }
  }
  // merge the two half-lanes holding the same column
  float m2 = __shfl_xor(m, 16, 32);
  float s2 = __shfl_xor(s, 16, 32);
  float mn = fmaxf(m, m2);
  float sm = s * __expf(m - mn) + s2 * __expf(m2 - mn);
  if (l < 16) {
    cmax[(size_t)bh * TT + kcol] = mn;
    csum[(size_t)bh * TT + kcol] = sm;
  }
}

// ---------------- v' = v / Z, stored transposed [B,H,HS,T] ------------------
__global__ __launch_bounds__(256) void vscale(
    const _Float16* __restrict__ v, const float* __restrict__ csum,
    _Float16* __restrict__ vT) {
  long i = (long)blockIdx.x * 256 + threadIdx.x;
  if (i >= (long)BB * HH * TT * HS) return;
  int e = (int)(i & (HS - 1));
  int kt = (int)((i >> 6) & (TT - 1));
  int bh = (int)(i >> 17);
  float Z = csum[(size_t)bh * TT + kt];
  float val = (float)v[i] / Z;
  vT[(((size_t)bh * HS + e)) * TT + kt] = to_h(val);
}

// ---------------- attention pass 2: attn = P @ V' ---------------------------
__global__ __launch_bounds__(256) void attn_pass2(
    const _Float16* __restrict__ q, const _Float16* __restrict__ k,
    const _Float16* __restrict__ vT, const float* __restrict__ cmax,
    _Float16* __restrict__ attn) {
  __shared__ _Float16 pbuf[8][16 * PAD];   // per-wave P tile [16q x 32k]
  const int bh = blockIdx.y;
  const int wave = threadIdx.x >> 5, l = threadIdx.x & 31;
  const int q0 = blockIdx.x * 128 + wave * 16;   // this wave's 16 query rows
  const int bi = bh / HH, h = bh % HH;
  const _Float16* qh  = q  + (size_t)bh * TT * HS;
  const _Float16* kh  = k  + (size_t)bh * TT * HS;
  const _Float16* vTh = vT + (size_t)bh * HS * TT;
  const float*    mh  = cmax + (size_t)bh * TT;

  v16h qa0 = load_frag(qh + (size_t)q0 * HS, HS);
  v16h qa1 = load_frag(qh + (size_t)q0 * HS + 32, HS);

  v8f acc[4];
  for (int j = 0; j < 4; ++j) acc[j] = (v8f){};

  const int cn = l & 15, rb = (l >> 4) * 8;
  for (int kt = 0; kt < q0 + 16; kt += 32) {
    // build P tile (two 16-wide halves) in LDS, C-layout -> A-layout
    for (int h2 = 0; h2 < 2; ++h2) {
      int k0h = kt + 16 * h2;
      if (k0h < q0 + 16) {
        v16h kf0 = load_frag(kh + (size_t)k0h * HS, HS);
        v16h kf1 = load_frag(kh + (size_t)k0h * HS + 32, HS);
        v8f cc = (v8f){};
        cc = wmma16(qa0, kf0, cc);
        cc = wmma16(qa1, kf1, cc);
        int kcol = k0h + cn;
        float mk = mh[kcol];
        for (int r = 0; r < 8; ++r) {
          int qi = q0 + rb + r;
          float pv = (qi >= kcol) ? __expf(cc[r] - mk) : 0.0f;
          pbuf[wave][(size_t)(rb + r) * PAD + 16 * h2 + cn] = to_h(pv);
        }
      } else {
        for (int r = 0; r < 8; ++r)
          pbuf[wave][(size_t)(rb + r) * PAD + 16 * h2 + cn] = (_Float16)0.0f;
      }
    }
    // intra-wave LDS turnaround (no block barrier: waves have different trip counts)
    asm volatile("s_wait_dscnt 0" ::: "memory");
    v16h pa = load_frag(pbuf[wave], PAD);
    for (int j = 0; j < 4; ++j) {
      v16h vb = load_frag(vTh + (size_t)(16 * j) * TT + kt, TT);
      acc[j] = wmma16(pa, vb, acc[j]);
    }
  }

  // write attn as f16 [B,T,D] (feeds the proj GEMM)
  for (int j = 0; j < 4; ++j)
    for (int r = 0; r < 8; ++r) {
      int qi = q0 + rb + r;
      int d = h * HS + 16 * j + cn;
      attn[((size_t)bi * TT + qi) * DD + d] = to_h(acc[j][r]);
    }
}

// ---------------- host-side orchestration ----------------
extern "C" void kernel_launch(void* const* d_in, const int* in_sizes, int n_in,
                              void* d_out, int out_size, void* d_ws,
                              size_t ws_size, hipStream_t stream) {
  const float* x     = (const float*)d_in[0];
  const float* Wq    = (const float*)d_in[1];
  const float* Wk    = (const float*)d_in[2];
  const float* Wv    = (const float*)d_in[3];
  const float* Wproj = (const float*)d_in[4];
  const float* bproj = (const float*)d_in[5];
  const float* W1    = (const float*)d_in[6];
  const float* b1    = (const float*)d_in[7];
  const float* W2    = (const float*)d_in[8];
  const float* b2    = (const float*)d_in[9];
  const float* ln1g  = (const float*)d_in[10];
  const float* ln1b  = (const float*)d_in[11];
  const float* ln2g  = (const float*)d_in[12];
  const float* ln2b  = (const float*)d_in[13];
  float* out = (float*)d_out;

  char* base = (char*)d_ws;
  size_t off = 0;
  auto alloc = [&](size_t bytes) -> char* {
    char* p = base + off;
    off += (bytes + 255) & ~(size_t)255;
    return p;
  };
  _Float16* wqkvT = (_Float16*)alloc((size_t)3 * DD * DD * 2);  // [3D][D]
  _Float16* wpT   = (_Float16*)alloc((size_t)DD * DD * 2);      // [D][D]
  _Float16* w1T   = (_Float16*)alloc((size_t)DFF * DD * 2);     // [4D][D]
  _Float16* w2T   = (_Float16*)alloc((size_t)DD * DFF * 2);     // [D][4D]
  float*    xn_f  = (float*)   alloc((size_t)BT * DD * 4);
  _Float16* xn_h  = (_Float16*)alloc((size_t)BT * DD * 2);
  _Float16* q_h   = (_Float16*)alloc((size_t)BB * HH * TT * HS * 2);
  _Float16* k_h   = (_Float16*)alloc((size_t)BB * HH * TT * HS * 2);
  _Float16* v_h   = (_Float16*)alloc((size_t)BB * HH * TT * HS * 2);
  _Float16* vT_h  = (_Float16*)alloc((size_t)BB * HH * TT * HS * 2);
  float*    cmax  = (float*)   alloc((size_t)BB * HH * TT * 4);
  float*    csum  = (float*)   alloc((size_t)BB * HH * TT * 4);
  _Float16* attnh = (_Float16*)alloc((size_t)BT * DD * 2);
  float*    x1_f  = (float*)   alloc((size_t)BT * DD * 4);
  float*    x1n_f = (float*)   alloc((size_t)BT * DD * 4);
  _Float16* x1n_h = (_Float16*)alloc((size_t)BT * DD * 2);
  _Float16* h_h   = (_Float16*)alloc((size_t)BT * DFF * 2);

  // 1) pack all weights to f16, transposed [N][K]
  {
    long tot = (long)3 * DD * DD + (long)DD * DD + 2L * DD * DFF;
    pack_weights<<<(unsigned)((tot + 255) / 256), 256, 0, stream>>>(
        Wq, Wk, Wv, Wproj, W1, W2, wqkvT, wpT, w1T, w2T);
  }
  // 2) LN1 -> xn (f32 + f16)
  ln_kernel<<<BT, 256, 0, stream>>>(x, ln1g, ln1b, xn_f, xn_h);
  // 3) QKV GEMM: [4096x1024] x [1024x3072]
  gemm128<<<dim3(3 * DD / 128, BT / 128), 256, 0, stream>>>(
      xn_h, wqkvT, 3 * DD, DD, MODE_QKV, nullptr, nullptr, nullptr, nullptr,
      q_h, k_h, v_h);
  // 4) per-column softmax stats
  attn_pass1<<<dim3(TT / 128, BB * HH), 256, 0, stream>>>(q_h, k_h, cmax, csum);
  // 5) v' = v / Z, transposed
  vscale<<<(unsigned)(((long)BB * HH * TT * HS + 255) / 256), 256, 0, stream>>>(
      v_h, csum, vT_h);
  // 6) attn = P @ V'
  attn_pass2<<<dim3(TT / 128, BB * HH), 256, 0, stream>>>(q_h, k_h, vT_h, cmax,
                                                          attnh);
  // 7) proj: x1 = xn + attn @ Wproj + bproj
  gemm128<<<dim3(DD / 128, BT / 128), 256, 0, stream>>>(
      attnh, wpT, DD, DD, MODE_PROJ, bproj, xn_f, x1_f, nullptr, nullptr,
      nullptr, nullptr);
  // 8) LN2 -> x1n (f32 + f16)
  ln_kernel<<<BT, 256, 0, stream>>>(x1_f, ln2g, ln2b, x1n_f, x1n_h);
  // 9) FFN1: h = relu(x1n @ W1 + b1)
  gemm128<<<dim3(DFF / 128, BT / 128), 256, 0, stream>>>(
      x1n_h, w1T, DFF, DD, MODE_FFN1, b1, nullptr, nullptr, h_h, nullptr,
      nullptr, nullptr);
  // 10) FFN2: out = x1n + h @ W2 + b2
  gemm128<<<dim3(DD / 128, BT / 128), 256, 0, stream>>>(
      h_h, w2T, DD, DFF, MODE_FFN2, b2, x1n_f, out, nullptr, nullptr, nullptr,
      nullptr);
}
